// Coarse_Grain_3DLayer_10179072491647
// MI455X (gfx1250) — matrix operand order repaired and checked by reading, not compiled, not run on hardware
//
#include <hip/hip_runtime.h>
#include <hip/hip_bf16.h>

#define N_NODES 8000
#define N_EDGES 128000
#define HDIM 256
#define EFDIM 64
#define EIN 591          // 2H + EF + 15
#define EINP 608         // padded to multiple of 32

typedef __bf16 bf16_t;
typedef __attribute__((ext_vector_type(16))) __bf16 bf16x16;
typedef __attribute__((ext_vector_type(8)))  float  f32x8;

__device__ __forceinline__ float lrelu_f(float x) { return x >= 0.f ? x : 0.01f * x; }

__device__ __forceinline__ f32x8 wmma_bf16(bf16x16 a, bf16x16 b, f32x8 c) {
  return __builtin_amdgcn_wmma_f32_16x16x32_bf16(false, a, false, b, (short)0, c, false, false);
}

// A-fragment (16x32 bf16, M x K): lane holds row (lane&15), k-base 16*(lane>>4);
// elements 0..15 at k..k+15 contiguous -> two 16B loads.
__device__ __forceinline__ bf16x16 frag_a_lds(const bf16_t* tile16x32, int lane) {
  const bf16_t* p = tile16x32 + (lane & 15) * 32 + ((lane >> 4) * 16);
  bf16x16 f;
  ((uint4*)&f)[0] = *(const uint4*)(p);
  ((uint4*)&f)[1] = *(const uint4*)(p + 8);
  return f;
}
// B-fragment (32x16 bf16, K x N) from [n][k] layout: lane n = lane&15, khalf = 16*(lane>>4);
// 16 contiguous k values -> two 16B loads.
__device__ __forceinline__ bf16x16 frag_b_contig(const bf16_t* p /* at [n][khalf] */) {
  bf16x16 f;
  ((uint4*)&f)[0] = *(const uint4*)(p);
  ((uint4*)&f)[1] = *(const uint4*)(p + 8);
  return f;
}

// ---------------------------------------------------------------------------
// Generic GEMM: C[M,Nc] = A[M,K](bf16) @ W[K,Nc](f32->bf16) + bias
// Tile: 128x64, K-step 32. 8 waves, each 32x32 (2x2 wmma).
// ---------------------------------------------------------------------------
__global__ __launch_bounds__(256) void gemm_bf16_kernel(
    const bf16_t* __restrict__ A, const float* __restrict__ W,
    const float* __restrict__ bias, float* __restrict__ C,
    int M, int K, int Nc) {
  __shared__ bf16_t As[128 * 32];
  __shared__ bf16_t Bs[64 * 32];   // transposed: [n][k]
  const int tid = threadIdx.x;
  const int lane = tid & 31, wave = tid >> 5;
  const int bm = blockIdx.x * 128;
  const int bn = blockIdx.y * 64;
  const int wm = (wave >> 1) * 32, wn = (wave & 1) * 32;
  f32x8 acc[2][2] = {};
  const int ar = tid >> 1, ac = (tid & 1) * 16;      // A staging: 16 elems/thread
  const int bnl = tid >> 2, bkk = (tid & 3) * 8;     // B staging: 8 elems/thread

  for (int k0 = 0; k0 < K; k0 += 32) {
    __syncthreads();
    {
      uint4* d4 = (uint4*)(As + ar * 32 + ac);
      int gr = bm + ar;
      if (gr < M) {
        const bf16_t* src = A + (size_t)gr * K + k0 + ac;
        d4[0] = ((const uint4*)src)[0];
        d4[1] = ((const uint4*)src)[1];
      } else {
        uint4 z = make_uint4(0u, 0u, 0u, 0u);
        d4[0] = z; d4[1] = z;
      }
    }
    {
      bf16_t* dst = Bs + bnl * 32 + bkk;
      const float* src = W + (size_t)(k0 + bkk) * Nc + bn + bnl;
      #pragma unroll
      for (int i = 0; i < 8; ++i) dst[i] = (bf16_t)src[(size_t)i * Nc];
    }
    __syncthreads();
    #pragma unroll
    for (int mi = 0; mi < 2; ++mi) {
      bf16x16 af = frag_a_lds(As + (wm + mi * 16) * 32, lane);
      #pragma unroll
      for (int ni = 0; ni < 2; ++ni) {
        bf16x16 bfr = frag_b_contig(Bs + (wn + ni * 16 + (lane & 15)) * 32 + ((lane >> 4) * 16));
        acc[mi][ni] = wmma_bf16(af, bfr, acc[mi][ni]);
      }
    }
  }
  const int h = lane >> 4, nl = lane & 15;
  #pragma unroll
  for (int mi = 0; mi < 2; ++mi)
    #pragma unroll
    for (int ni = 0; ni < 2; ++ni) {
      int col = bn + wn + ni * 16 + nl;
      float bb = bias ? bias[col] : 0.f;
      #pragma unroll
      for (int r = 0; r < 8; ++r) {
        int row = bm + wm + mi * 16 + r + 8 * h;
        if (row < M) C[(size_t)row * Nc + col] = acc[mi][ni][r] + bb;
      }
    }
}

// ---------------------------------------------------------------------------
// Row LayerNorm over 256 cols: y = LN(a0*X + a1*skip)*g + b, optional lrelu.
// One wave per row, 8 rows per block. Optional f32 and/or bf16 outputs.
// ---------------------------------------------------------------------------
__global__ __launch_bounds__(256) void ln_kernel(
    const float* __restrict__ X, const float* __restrict__ skip,
    float a0, float a1, const float* __restrict__ g, const float* __restrict__ b,
    float* __restrict__ outf, bf16_t* __restrict__ outb, int M, int do_lrelu) {
  int row = blockIdx.x * 8 + (threadIdx.x >> 5);
  if (row >= M) return;
  int lane = threadIdx.x & 31;
  const float* xr = X + (size_t)row * HDIM;
  float v[8];
  float s = 0.f, ss = 0.f;
  #pragma unroll
  for (int i = 0; i < 8; ++i) {
    int c = lane + i * 32;
    float val = a0 * xr[c];
    if (skip) val += a1 * skip[(size_t)row * HDIM + c];
    v[i] = val; s += val; ss += val * val;
  }
  #pragma unroll
  for (int off = 16; off > 0; off >>= 1) {
    s += __shfl_xor(s, off, 32);
    ss += __shfl_xor(ss, off, 32);
  }
  float mean = s * (1.f / HDIM);
  float var = ss * (1.f / HDIM) - mean * mean;
  float rs = rsqrtf(var + 1e-5f);
  #pragma unroll
  for (int i = 0; i < 8; ++i) {
    int c = lane + i * 32;
    float y = (v[i] - mean) * rs * g[c] + b[c];
    if (do_lrelu) y = lrelu_f(y);
    if (outf) outf[(size_t)row * HDIM + c] = y;
    if (outb) outb[(size_t)row * HDIM + c] = (bf16_t)y;
  }
}

// ---------------------------------------------------------------------------
// Flash attention: out = softmax(mask*(Q K^T) - 1000*(1-mask)) @ V
// qb: [N,256] bf16, kb: [N,256] bf16, vt: [256,N] bf16 (transposed),
// mask f32 [N,N]; maskT selects mask[col][row]. Block: 8 waves x 16 q-rows.
// ---------------------------------------------------------------------------
__global__ __launch_bounds__(256) void attn_kernel(
    const bf16_t* __restrict__ qb, const bf16_t* __restrict__ kb,
    const bf16_t* __restrict__ vt, const float* __restrict__ mask,
    int maskT, float* __restrict__ out, int Nn) {
  __shared__ bf16_t Plds[8][16 * 32];
  const int lane = threadIdx.x & 31, wave = threadIdx.x >> 5;
  const int qbase = blockIdx.x * 128 + wave * 16;
  if (qbase >= Nn) return;
  const int h = lane >> 4, nl = lane & 15;

  // Preload Q fragments (16 rows x 256 dims = 8 chunks)
  bf16x16 qf[8];
  {
    const bf16_t* base = qb + (size_t)(qbase + nl) * HDIM + h * 16;
    #pragma unroll
    for (int d = 0; d < 8; ++d) {
      const bf16_t* p = base + d * 32;
      ((uint4*)&qf[d])[0] = *(const uint4*)(p);
      ((uint4*)&qf[d])[1] = *(const uint4*)(p + 8);
    }
  }
  f32x8 oacc[16] = {};
  float mrun[8], lrun[8];
  #pragma unroll
  for (int r = 0; r < 8; ++r) { mrun[r] = -1e30f; lrun[r] = 0.f; }

  for (int kt = 0; kt < Nn; kt += 32) {
    f32x8 s0 = {}, s1 = {};
    {
      const bf16_t* pk0 = kb + (size_t)(kt + nl) * HDIM + h * 16;
      const bf16_t* pk1 = kb + (size_t)(kt + 16 + nl) * HDIM + h * 16;
      #pragma unroll
      for (int d = 0; d < 8; ++d) {
        s0 = wmma_bf16(qf[d], frag_b_contig(pk0 + d * 32), s0);
        s1 = wmma_bf16(qf[d], frag_b_contig(pk1 + d * 32), s1);
      }
    }
    float p0v[8], p1v[8];
    #pragma unroll
    for (int r = 0; r < 8; ++r) {
      int row = qbase + r + 8 * h;
      int c0 = kt + nl, c1 = c0 + 16;
      float m0 = maskT ? mask[(size_t)c0 * Nn + row] : mask[(size_t)row * Nn + c0];
      float m1 = maskT ? mask[(size_t)c1 * Nn + row] : mask[(size_t)row * Nn + c1];
      float sv0 = m0 * s0[r] - 1000.f * (1.f - m0);
      float sv1 = m1 * s1[r] - 1000.f * (1.f - m1);
      float mx = fmaxf(sv0, sv1);
      #pragma unroll
      for (int off = 1; off < 16; off <<= 1) mx = fmaxf(mx, __shfl_xor(mx, off, 32));
      float mnew = fmaxf(mrun[r], mx);
      float alpha = __expf(mrun[r] - mnew);
      float e0 = __expf(sv0 - mnew), e1 = __expf(sv1 - mnew);
      float ls = e0 + e1;
      #pragma unroll
      for (int off = 1; off < 16; off <<= 1) ls += __shfl_xor(ls, off, 32);
      lrun[r] = lrun[r] * alpha + ls;
      mrun[r] = mnew;
      p0v[r] = e0; p1v[r] = e1;
      #pragma unroll
      for (int nc = 0; nc < 16; ++nc) oacc[nc][r] *= alpha;
    }
    // C-layout -> A-layout for P via per-wave LDS staging
    bf16_t* P = Plds[wave];
    #pragma unroll
    for (int r = 0; r < 8; ++r) {
      P[(r + 8 * h) * 32 + nl]      = (bf16_t)p0v[r];
      P[(r + 8 * h) * 32 + 16 + nl] = (bf16_t)p1v[r];
    }
    bf16x16 pf = frag_a_lds(P, lane);
    {
      const int kh = h * 16;
      #pragma unroll
      for (int nc = 0; nc < 16; ++nc) {
        const bf16_t* pv = vt + (size_t)(nc * 16 + nl) * Nn + kt + kh;
        oacc[nc] = wmma_bf16(pf, frag_b_contig(pv), oacc[nc]);
      }
    }
  }
  #pragma unroll
  for (int nc = 0; nc < 16; ++nc)
    #pragma unroll
    for (int r = 0; r < 8; ++r) {
      int row = qbase + r + 8 * h;
      out[(size_t)row * HDIM + nc * 16 + nl] = oacc[nc][r] / lrun[r];
    }
}

// ---------------------------------------------------------------------------
// Small helper kernels
// ---------------------------------------------------------------------------
__global__ void pad_we1_kernel(const float* __restrict__ W, float* __restrict__ Wp) {
  int idx = blockIdx.x * 256 + threadIdx.x;
  if (idx >= EINP * HDIM) return;
  int row = idx / HDIM;
  Wp[idx] = (row < EIN) ? W[idx] : 0.f;
}

__global__ void lrelu_cvt_kernel(const float* __restrict__ X, bf16_t* __restrict__ Y, size_t n) {
  size_t i = (size_t)blockIdx.x * blockDim.x + threadIdx.x;
  if (i < n) Y[i] = (bf16_t)lrelu_f(X[i]);
}

__global__ void transpose_cvt_kernel(const float* __restrict__ V, bf16_t* __restrict__ Vt, int Nn) {
  size_t idx = (size_t)blockIdx.x * blockDim.x + threadIdx.x;
  if (idx >= (size_t)Nn * HDIM) return;
  int row = (int)(idx / HDIM), col = (int)(idx % HDIM);
  Vt[(size_t)col * Nn + row] = (bf16_t)V[idx];
}

// Build edge-MLP input: [h[src], h[dst], efeat, rbf(dist), zero-pad] as bf16, plus x_rel.
__global__ __launch_bounds__(256) void edge_concat_kernel(
    const float* __restrict__ hh, const float* __restrict__ x,
    const float* __restrict__ efeat, const int* __restrict__ src,
    const int* __restrict__ dst, bf16_t* __restrict__ Min,
    float* __restrict__ x_rel, int E_) {
  int e = blockIdx.x * 8 + (threadIdx.x >> 5);
  if (e >= E_) return;
  int lane = threadIdx.x & 31;
  int s = src[e], d = dst[e];
  float xr0 = x[s * 3 + 0] - x[d * 3 + 0];
  float xr1 = x[s * 3 + 1] - x[d * 3 + 1];
  float xr2 = x[s * 3 + 2] - x[d * 3 + 2];
  float d2 = xr0 * xr0 + xr1 * xr1 + xr2 * xr2;
  if (lane == 0) {
    x_rel[(size_t)e * 3 + 0] = xr0;
    x_rel[(size_t)e * 3 + 1] = xr1;
    x_rel[(size_t)e * 3 + 2] = xr2;
  }
  bf16_t* mrow = Min + (size_t)e * EINP;
  for (int i = lane; i < EINP; i += 32) {
    float val;
    if (i < HDIM)            val = hh[(size_t)s * HDIM + i];
    else if (i < 2 * HDIM)   val = hh[(size_t)d * HDIM + (i - HDIM)];
    else if (i < 2 * HDIM + EFDIM) val = efeat[(size_t)e * EFDIM + (i - 2 * HDIM)];
    else if (i < EIN) {
      int j = i - (2 * HDIM + EFDIM);
      val = __expf(-d2 / __powf(1.5f, (float)j));
    } else val = 0.f;
    mrow[i] = (bf16_t)val;
  }
}

__global__ void seg_count_kernel(const int* __restrict__ dst, float* __restrict__ counts, int E_) {
  int e = blockIdx.x * 256 + threadIdx.x;
  if (e < E_) atomicAdd(&counts[dst[e]], 1.f);
}

__global__ __launch_bounds__(256) void seg_sum_h_kernel(
    const float* __restrict__ msg, const int* __restrict__ dst,
    float* __restrict__ aggr, int E_) {
  int e = blockIdx.x * 8 + (threadIdx.x >> 5);
  if (e >= E_) return;
  int lane = threadIdx.x & 31;
  int d = dst[e];
  for (int i = lane; i < HDIM; i += 32)
    atomicAdd(&aggr[(size_t)d * HDIM + i], msg[(size_t)e * HDIM + i]);
}

__global__ void div_aggr_kernel(float* __restrict__ aggr, const float* __restrict__ counts, int Nn) {
  size_t i = (size_t)blockIdx.x * blockDim.x + threadIdx.x;
  if (i < (size_t)Nn * HDIM) aggr[i] /= fmaxf(counts[i / HDIM], 1.f);
}

__global__ __launch_bounds__(256) void coef_kernel(
    const float* __restrict__ Cin, const float* __restrict__ Wc2,
    const float* __restrict__ bc2, float* __restrict__ coef, int E_) {
  int e = blockIdx.x * 8 + (threadIdx.x >> 5);
  if (e >= E_) return;
  int lane = threadIdx.x & 31;
  const float* row = Cin + (size_t)e * HDIM;
  float s = 0.f;
  for (int i = lane; i < HDIM; i += 32) s += row[i] * Wc2[i];
  #pragma unroll
  for (int off = 16; off > 0; off >>= 1) s += __shfl_xor(s, off, 32);
  if (lane == 0) coef[e] = s + bc2[0];
}

__global__ void seg_sum_x_kernel(const float* __restrict__ x_rel, const float* __restrict__ coef,
                                 const int* __restrict__ dst, float* __restrict__ xupd, int E_) {
  int e = blockIdx.x * 256 + threadIdx.x;
  if (e >= E_) return;
  float cf = coef[e];
  int d = dst[e];
  #pragma unroll
  for (int j = 0; j < 3; ++j)
    atomicAdd(&xupd[(size_t)d * 3 + j], x_rel[(size_t)e * 3 + j] * cf);
}

__global__ void finalize_x_kernel(const float* __restrict__ orig_x, const float* __restrict__ x,
                                  const float* __restrict__ xupd, const float* __restrict__ counts,
                                  float* __restrict__ xout, int Nn) {
  int i = blockIdx.x * 256 + threadIdx.x;
  if (i >= Nn * 3) return;
  int node = i / 3;
  xout[i] = 0.25f * orig_x[i] + 0.75f * x[i] + xupd[i] / fmaxf(counts[node], 1.f);
}

__global__ void node_concat_kernel(const float* __restrict__ hh, const float* __restrict__ aggr,
                                   const float* __restrict__ cross, bf16_t* __restrict__ ni, int Nn) {
  size_t idx = (size_t)blockIdx.x * blockDim.x + threadIdx.x;
  if (idx >= (size_t)Nn * 768) return;
  int row = (int)(idx / 768), col = (int)(idx % 768);
  float v;
  if (col < HDIM)          v = hh[(size_t)row * HDIM + col];
  else if (col < 2 * HDIM) v = aggr[(size_t)row * HDIM + (col - HDIM)];
  else                     v = cross[(size_t)row * HDIM + (col - 2 * HDIM)];
  ni[idx] = (bf16_t)v;
}

// ---------------------------------------------------------------------------
extern "C" void kernel_launch(void* const* d_in, const int* in_sizes, int n_in,
                              void* d_out, int out_size, void* d_ws, size_t ws_size,
                              hipStream_t stream) {
  (void)in_sizes; (void)n_in; (void)out_size; (void)ws_size;
  const int N = N_NODES, E = N_EDGES;

  const float* h_A  = (const float*)d_in[0];
  const float* x_A  = (const float*)d_in[1];
  const float* ox_A = (const float*)d_in[2];
  const float* ef_A = (const float*)d_in[3];
  const int*   ei_A = (const int*)d_in[4];
  const float* h_B  = (const float*)d_in[5];
  const float* x_B  = (const float*)d_in[6];
  const float* ox_B = (const float*)d_in[7];
  const float* ef_B = (const float*)d_in[8];
  const int*   ei_B = (const int*)d_in[9];
  const float* mask = (const float*)d_in[10];
  const float* We1 = (const float*)d_in[11]; const float* be1 = (const float*)d_in[12];
  const float* ge1 = (const float*)d_in[13]; const float* bb1 = (const float*)d_in[14];
  const float* We2 = (const float*)d_in[15]; const float* be2 = (const float*)d_in[16];
  const float* ge2 = (const float*)d_in[17]; const float* bb2 = (const float*)d_in[18];
  const float* Wq  = (const float*)d_in[19]; const float* Wk  = (const float*)d_in[20];
  const float* Wv  = (const float*)d_in[21];
  const float* Wn1 = (const float*)d_in[22]; const float* bn1 = (const float*)d_in[23];
  const float* gn1 = (const float*)d_in[24]; const float* bbn1= (const float*)d_in[25];
  const float* Wn2 = (const float*)d_in[26]; const float* bn2 = (const float*)d_in[27];
  const float* gn2 = (const float*)d_in[28]; const float* bbn2= (const float*)d_in[29];
  const float* Wc1 = (const float*)d_in[30]; const float* bc1 = (const float*)d_in[31];
  const float* gc1 = (const float*)d_in[32]; const float* bbc1= (const float*)d_in[33];
  const float* Wc2 = (const float*)d_in[34]; const float* bc2 = (const float*)d_in[35];
  const float* gf  = (const float*)d_in[36]; const float* bf_ = (const float*)d_in[37];

  float* out = (float*)d_out;
  float* out_hA = out;
  float* out_xA = out + (size_t)N * HDIM;
  float* out_hB = out + (size_t)N * HDIM + (size_t)N * 3;
  float* out_xB = out + 2 * (size_t)N * HDIM + (size_t)N * 3;

  // ---- workspace carve-up ----
  uint8_t* wsp = (uint8_t*)d_ws;
  size_t off = 0;
  auto alloc = [&](size_t bytes) -> void* {
    void* p = wsp + off;
    off = (off + bytes + 255) & ~(size_t)255;
    return p;
  };
  bf16_t* Minb  = (bf16_t*)alloc((size_t)E * EINP * sizeof(bf16_t));
  float*  tE1   = (float*) alloc((size_t)E * HDIM * sizeof(float));
  bf16_t* Eb    = (bf16_t*)alloc((size_t)E * HDIM * sizeof(bf16_t));
  float*  xrel  = (float*) alloc((size_t)E * 3 * sizeof(float));
  float*  coef  = (float*) alloc((size_t)E * sizeof(float));
  float*  aggr  = (float*) alloc((size_t)N * HDIM * sizeof(float));
  float*  counts= (float*) alloc((size_t)N * sizeof(float));
  float*  xupd  = (float*) alloc((size_t)N * 3 * sizeof(float));
  bf16_t* hnAb  = (bf16_t*)alloc((size_t)N * HDIM * sizeof(bf16_t));
  bf16_t* hnBb  = (bf16_t*)alloc((size_t)N * HDIM * sizeof(bf16_t));
  bf16_t* qAb   = (bf16_t*)alloc((size_t)N * HDIM * sizeof(bf16_t));
  bf16_t* kAb   = (bf16_t*)alloc((size_t)N * HDIM * sizeof(bf16_t));
  bf16_t* qBb   = (bf16_t*)alloc((size_t)N * HDIM * sizeof(bf16_t));
  bf16_t* kBb   = (bf16_t*)alloc((size_t)N * HDIM * sizeof(bf16_t));
  bf16_t* VtA   = (bf16_t*)alloc((size_t)HDIM * N * sizeof(bf16_t));
  bf16_t* VtB   = (bf16_t*)alloc((size_t)HDIM * N * sizeof(bf16_t));
  float*  crossA= (float*) alloc((size_t)N * HDIM * sizeof(float));
  float*  crossB= (float*) alloc((size_t)N * HDIM * sizeof(float));
  float*  tN    = (float*) alloc((size_t)N * HDIM * sizeof(float));
  bf16_t* niB   = (bf16_t*)alloc((size_t)N * 768 * sizeof(bf16_t));
  bf16_t* u1b   = (bf16_t*)alloc((size_t)N * HDIM * sizeof(bf16_t));
  float*  We1p  = (float*) alloc((size_t)EINP * HDIM * sizeof(float));

  auto gemm = [&](const bf16_t* A, const float* W, const float* bias, float* C,
                  int M, int K, int Nc) {
    dim3 g((M + 127) / 128, Nc / 64);
    gemm_bf16_kernel<<<g, 256, 0, stream>>>(A, W, bias, C, M, K, Nc);
  };
  auto lnk = [&](const float* X, const float* skip, float a0, float a1,
                 const float* g_, const float* b_, float* outf, bf16_t* outb,
                 int M, int dolr) {
    ln_kernel<<<(M + 7) / 8, 256, 0, stream>>>(X, skip, a0, a1, g_, b_, outf, outb, M, dolr);
  };

  // ---- prep ----
  pad_we1_kernel<<<(EINP * HDIM + 255) / 256, 256, 0, stream>>>(We1, We1p);
  lnk(h_A, nullptr, 1.f, 0.f, gf, bf_, nullptr, hnAb, N, 0);
  lnk(h_B, nullptr, 1.f, 0.f, gf, bf_, nullptr, hnBb, N, 0);

  const size_t NH = (size_t)N * HDIM;
  // projections (A side)
  gemm(hnAb, Wq, nullptr, tN, N, HDIM, HDIM);
  lrelu_cvt_kernel<<<(int)((NH + 255) / 256), 256, 0, stream>>>(tN, qAb, NH);
  gemm(hnAb, Wk, nullptr, tN, N, HDIM, HDIM);
  lrelu_cvt_kernel<<<(int)((NH + 255) / 256), 256, 0, stream>>>(tN, kAb, NH);
  gemm(hnAb, Wv, nullptr, tN, N, HDIM, HDIM);
  transpose_cvt_kernel<<<(int)((NH + 255) / 256), 256, 0, stream>>>(tN, VtA, N);
  // projections (B side)
  gemm(hnBb, Wq, nullptr, tN, N, HDIM, HDIM);
  lrelu_cvt_kernel<<<(int)((NH + 255) / 256), 256, 0, stream>>>(tN, qBb, NH);
  gemm(hnBb, Wk, nullptr, tN, N, HDIM, HDIM);
  lrelu_cvt_kernel<<<(int)((NH + 255) / 256), 256, 0, stream>>>(tN, kBb, NH);
  gemm(hnBb, Wv, nullptr, tN, N, HDIM, HDIM);
  transpose_cvt_kernel<<<(int)((NH + 255) / 256), 256, 0, stream>>>(tN, VtB, N);

  // ---- cross attention ----
  attn_kernel<<<(N + 127) / 128, 256, 0, stream>>>(qAb, kBb, VtB, mask, 0, crossA, N);
  attn_kernel<<<(N + 127) / 128, 256, 0, stream>>>(qBb, kAb, VtA, mask, 1, crossB, N);
  lnk(crossA, nullptr, 1.f, 0.f, gf, bf_, crossA, nullptr, N, 0);
  lnk(crossB, nullptr, 1.f, 0.f, gf, bf_, crossB, nullptr, N, 0);

  // ---- per-graph pipeline ----
  auto per_graph = [&](const float* hh, const float* xx, const float* oxx,
                       const float* ef, const int* src, const int* dst,
                       const float* cross, float* out_h, float* out_x) {
    edge_concat_kernel<<<(E + 7) / 8, 256, 0, stream>>>(hh, xx, ef, src, dst, Minb, xrel, E);
    gemm(Minb, We1p, be1, tE1, E, EINP, HDIM);
    lnk(tE1, nullptr, 1.f, 0.f, ge1, bb1, nullptr, Eb, E, 1);
    gemm(Eb, We2, be2, tE1, E, HDIM, HDIM);
    lnk(tE1, nullptr, 1.f, 0.f, ge2, bb2, tE1, Eb, E, 0);   // msg (f32 in-place + bf16)
    hipMemsetAsync(aggr, 0, (size_t)N * HDIM * sizeof(float), stream);
    hipMemsetAsync(counts, 0, (size_t)N * sizeof(float), stream);
    hipMemsetAsync(xupd, 0, (size_t)N * 3 * sizeof(float), stream);
    seg_count_kernel<<<(E + 255) / 256, 256, 0, stream>>>(dst, counts, E);
    seg_sum_h_kernel<<<(E + 7) / 8, 256, 0, stream>>>(tE1, dst, aggr, E);
    div_aggr_kernel<<<(int)((NH + 255) / 256), 256, 0, stream>>>(aggr, counts, N);
    gemm(Eb, Wc1, bc1, tE1, E, HDIM, HDIM);
    lnk(tE1, nullptr, 1.f, 0.f, gc1, bbc1, tE1, nullptr, E, 1);  // c (in-place)
    coef_kernel<<<(E + 7) / 8, 256, 0, stream>>>(tE1, Wc2, bc2, coef, E);
    seg_sum_x_kernel<<<(E + 255) / 256, 256, 0, stream>>>(xrel, coef, dst, xupd, E);
    finalize_x_kernel<<<(N * 3 + 255) / 256, 256, 0, stream>>>(oxx, xx, xupd, counts, out_x, N);
    node_concat_kernel<<<(int)(((size_t)N * 768 + 255) / 256), 256, 0, stream>>>(hh, aggr, cross, niB, N);
    gemm(niB, Wn1, bn1, tN, N, 768, HDIM);
    lnk(tN, nullptr, 1.f, 0.f, gn1, bbn1, nullptr, u1b, N, 1);
    gemm(u1b, Wn2, bn2, tN, N, HDIM, HDIM);
    lnk(tN, nullptr, 1.f, 0.f, gn2, bbn2, tN, nullptr, N, 0);
    lnk(tN, hh, 0.5f, 0.5f, gf, bf_, out_h, nullptr, N, 0);
  };

  per_graph(h_A, x_A, ox_A, ef_A, ei_A, ei_A + E, crossA, out_hA, out_xA);
  per_graph(h_B, x_B, ox_B, ef_B, ei_B, ei_B + E, crossB, out_hB, out_xB);
}
